// dynedge_angle_xfeats_2396591751240
// MI455X (gfx1250) — matrix operand order repaired and checked by reading, not compiled, not run on hardware
//
#include <hip/hip_runtime.h>
#include <hip/hip_bf16.h>

typedef _Float16 v16h __attribute__((ext_vector_type(16)));
typedef _Float16 v8h  __attribute__((ext_vector_type(8)));
typedef float    v8f  __attribute__((ext_vector_type(8)));

#define NBATCH 64
#define NPER   256
#define KNN    8

__device__ __forceinline__ float lrelu(float v) { return v > 0.f ? v : 0.01f * v; }

// ---------------------------------------------------------------------------
// Pack W (K x N, f32, row-major) -> Wp (N x Kpad, f16), zero K-padding.
// B-fragment for (col, kk, half) is then two runs of 8 contiguous halves: two b128 loads.
// ---------------------------------------------------------------------------
__global__ __launch_bounds__(256) void pack_weight_kernel(
    const float* __restrict__ W, int K, int N, int Kpad,
    _Float16* __restrict__ Wp) {
  int t = blockIdx.x * 256 + threadIdx.x;
  if (t >= N * Kpad) return;
  int n = t / Kpad;
  int k = t - n * Kpad;
  Wp[t] = (_Float16)((k < K) ? W[(size_t)k * N + n] : 0.f);
}

// ---------------------------------------------------------------------------
// KNN: per batch, per node, 8 nearest neighbors by first-3-feature distance.
// ---------------------------------------------------------------------------
__global__ __launch_bounds__(256) void knn_kernel(const float* __restrict__ fin,
                                                  int Sin, int* __restrict__ nidx) {
  const int b = blockIdx.x;
  const int i = threadIdx.x;
  const float* base = fin + (size_t)(b * NPER) * Sin;
  const float xi0 = base[i * Sin + 0];
  const float xi1 = base[i * Sin + 1];
  const float xi2 = base[i * Sin + 2];
  float bd[KNN];
  int   bj[KNN];
#pragma unroll
  for (int k = 0; k < KNN; ++k) { bd[k] = 3.0e38f; bj[k] = 0; }
  for (int j = 0; j < NPER; ++j) {
    if (j == i) continue;  // reference adds eye*1e10 -> self excluded
    float dx = base[j * Sin + 0] - xi0;
    float dy = base[j * Sin + 1] - xi1;
    float dz = base[j * Sin + 2] - xi2;
    float d2 = dx * dx + dy * dy + dz * dz;
    int jj = j;
#pragma unroll
    for (int p = 0; p < KNN; ++p) {  // bubble insert, stable (lower idx first)
      if (d2 < bd[p]) {
        float td = bd[p]; int tj = bj[p];
        bd[p] = d2; bj[p] = jj;
        d2 = td; jj = tj;
      }
    }
  }
#pragma unroll
  for (int k = 0; k < KNN; ++k)
    nidx[(((b << 8) + i) << 3) + k] = bj[k];
}

// ---------------------------------------------------------------------------
// Homophily on first 4 features of x, using all KNN indices.
// ---------------------------------------------------------------------------
__global__ __launch_bounds__(256) void homophily_kernel(const float* __restrict__ x,
                                                        const int* __restrict__ nidx,
                                                        float* __restrict__ h) {
  __shared__ float s[4];
  const int b = blockIdx.x;
  const int i = threadIdx.x;
  if (i < 4) s[i] = 0.f;
  __syncthreads();
  const float* xi = x + (size_t)((b << 8) + i) * 7;
  float c0 = 0.f, c1 = 0.f, c2 = 0.f, c3 = 0.f;
#pragma unroll
  for (int k = 0; k < KNN; ++k) {
    int j = nidx[(((b << 8) + i) << 3) + k];
    const float* xj = x + (size_t)((b << 8) + j) * 7;
    c0 += (xj[0] == xi[0]) ? 1.f : 0.f;
    c1 += (xj[1] == xi[1]) ? 1.f : 0.f;
    c2 += (xj[2] == xi[2]) ? 1.f : 0.f;
    c3 += (xj[3] == xi[3]) ? 1.f : 0.f;
  }
  atomicAdd(&s[0], c0); atomicAdd(&s[1], c1);
  atomicAdd(&s[2], c2); atomicAdd(&s[3], c3);
  __syncthreads();
  if (i < 4) h[b * 4 + i] = s[i] * (1.f / (float)(NPER * KNN));
}

// ---------------------------------------------------------------------------
// Fused EdgeConv: per block, 2 nodes (16 edges). Stage msg=[xi, xj-xi] tile in
// LDS (f16), GEMM1 (msg@Wa + ba, leaky) -> LDS, GEMM2 (h1@Wb + bb, leaky),
// sum over K=8 edges -> node output. Weights pre-packed [N][Kpad] f16.
// ---------------------------------------------------------------------------
__global__ __launch_bounds__(128) void edgeconv_kernel(
    const float* __restrict__ fin, int Sin, int Fin, int K1pad,
    const int* __restrict__ nidx,
    const _Float16* __restrict__ Wpa, const float* __restrict__ ba,
    int N1, int K2pad,
    const _Float16* __restrict__ Wpb, const float* __restrict__ bb,
    float* __restrict__ fout) {
  __shared__ __align__(16) _Float16 sA[16 * 512];  // msg tile (K1pad <= 512)
  __shared__ __align__(16) _Float16 sH[16 * 352];  // h1 tile  (K2pad <= 352)
  __shared__ int sJ[16];

  const int tid = threadIdx.x;
  const int g0  = blockIdx.x * 2;  // first global node of this block
  const int b   = g0 >> 8;
  const int i0  = g0 & 255;

  if (tid < 16) {
    int i = i0 + (tid >> 3);
    sJ[tid] = nidx[(((b << 8) + i) << 3) + (tid & 7)];
  }
  __syncthreads();

  // Stage message tile (16 edges x K1pad) as f16. Consecutive tids hit
  // consecutive c within a row -> coalesced, wave-uniform branches.
  for (int t = tid; t < 16 * K1pad; t += 128) {
    int e = t / K1pad;
    int c = t - e * K1pad;
    int i = i0 + (e >> 3);
    const float* xi = fin + (size_t)((b << 8) + i) * Sin;
    float v = 0.f;
    if (c < Fin) {
      v = xi[c];
    } else if (c < 2 * Fin) {
      const float* xj = fin + (size_t)((b << 8) + sJ[e]) * Sin;
      v = xj[c - Fin] - xi[c - Fin];
    }
    sA[e * K1pad + c] = (_Float16)v;
  }
  // Zero the K-padding columns of sH (read by GEMM2).
  for (int t = tid; t < 16 * (K2pad - N1); t += 128) {
    int pc = K2pad - N1;
    int e  = t / pc;
    int c  = N1 + (t - e * pc);
    sH[e * K2pad + c] = (_Float16)0.f;
  }
  __syncthreads();

  const int wave = tid >> 5;
  const int lane = tid & 31;
  const int row  = lane & 15;
  const int aSel = (lane >> 4) << 3;   // A-fragment K half-select (0 or 8)
  const int bSel = (lane >> 4) << 3;   // B-fragment K half-select (0 or 8)
  const int rb   = (lane >> 4) << 3;   // D-fragment row base (0 or 8)

  // ---- GEMM1: msg(16 x K1) @ Wa + ba, leaky -> sH ----
  const int nT1 = N1 >> 4;  // N1 is a multiple of 16
  for (int nt = wave; nt < nT1; nt += 4) {
    const int col = (nt << 4) + row;
    const _Float16* aBase = &sA[row * K1pad + aSel];
    const _Float16* bBase = Wpa + (col * K1pad + bSel);
    v8f acc = {0.f, 0.f, 0.f, 0.f, 0.f, 0.f, 0.f, 0.f};
    for (int kk = 0; kk < K1pad; kk += 32) {
      v8h alo = *(const v8h*)(aBase + kk);
      v8h ahi = *(const v8h*)(aBase + kk + 16);
      v16h af = __builtin_shufflevector(alo, ahi, 0, 1, 2, 3, 4, 5, 6, 7, 8, 9,
                                        10, 11, 12, 13, 14, 15);
      v8h blo = *(const v8h*)(bBase + kk);
      v8h bhi = *(const v8h*)(bBase + kk + 16);
      v16h bf = __builtin_shufflevector(blo, bhi, 0, 1, 2, 3, 4, 5, 6, 7, 8, 9,
                                        10, 11, 12, 13, 14, 15);
      acc = __builtin_amdgcn_wmma_f32_16x16x32_f16(false, af, false, bf,
                                                   (short)0, acc, false, false);
    }
    const float bias = ba[col];
#pragma unroll
    for (int r = 0; r < 8; ++r) {
      float v = lrelu(acc[r] + bias);
      sH[(rb + r) * K2pad + col] = (_Float16)v;
    }
  }
  __syncthreads();

  // ---- GEMM2: h1(16 x N1) @ Wb + bb, leaky, sum over 8 edges ----
  for (int nt = wave; nt < 16; nt += 4) {
    const int col = (nt << 4) + row;
    const _Float16* aBase = &sH[row * K2pad + aSel];
    const _Float16* bBase = Wpb + (col * K2pad + bSel);
    v8f acc = {0.f, 0.f, 0.f, 0.f, 0.f, 0.f, 0.f, 0.f};
    for (int kk = 0; kk < K2pad; kk += 32) {
      v8h alo = *(const v8h*)(aBase + kk);
      v8h ahi = *(const v8h*)(aBase + kk + 16);
      v16h af = __builtin_shufflevector(alo, ahi, 0, 1, 2, 3, 4, 5, 6, 7, 8, 9,
                                        10, 11, 12, 13, 14, 15);
      v8h blo = *(const v8h*)(bBase + kk);
      v8h bhi = *(const v8h*)(bBase + kk + 16);
      v16h bf = __builtin_shufflevector(blo, bhi, 0, 1, 2, 3, 4, 5, 6, 7, 8, 9,
                                        10, 11, 12, 13, 14, 15);
      acc = __builtin_amdgcn_wmma_f32_16x16x32_f16(false, af, false, bf,
                                                   (short)0, acc, false, false);
    }
    const float bias = bb[col];
    float s = 0.f;
#pragma unroll
    for (int r = 0; r < 8; ++r) s += lrelu(acc[r] + bias);
    // lanes 0-15 hold edge-rows 0..7 (node 0), lanes 16-31 rows 8..15 (node 1)
    const int node = i0 + (lane >> 4);
    fout[(size_t)((b << 8) + node) * 256 + col] = s;
  }
}

// ---------------------------------------------------------------------------
// Fused per-node MLP: concat [x(7), a, b, c, d] (1031) @ Wn1(1031x336)+bn1
// leaky, then @ Wn2(336x256)+bn2 (no act). 16 nodes per block.
// ---------------------------------------------------------------------------
__global__ __launch_bounds__(128) void nodemlp_kernel(
    const float* __restrict__ x,  const float* __restrict__ fa,
    const float* __restrict__ fb, const float* __restrict__ fc,
    const float* __restrict__ fd,
    const _Float16* __restrict__ Wp1, const float* __restrict__ bn1,
    const _Float16* __restrict__ Wp2, const float* __restrict__ bn2,
    float* __restrict__ x2) {
  __shared__ __align__(16) _Float16 sA[16 * 1056];  // 1031 padded to 33*32
  __shared__ __align__(16) _Float16 sH[16 * 352];   // 336 padded to 11*32
  const int tid = threadIdx.x;
  const int g0  = blockIdx.x * 16;

  for (int t = tid; t < 16 * 1056; t += 128) {
    int e = t / 1056;
    int c = t - e * 1056;
    int node = g0 + e;
    float v = 0.f;
    if (c < 7)          v = x[(size_t)node * 7 + c];
    else if (c < 263)   v = fa[(size_t)node * 256 + (c - 7)];
    else if (c < 519)   v = fb[(size_t)node * 256 + (c - 263)];
    else if (c < 775)   v = fc[(size_t)node * 256 + (c - 519)];
    else if (c < 1031)  v = fd[(size_t)node * 256 + (c - 775)];
    sA[e * 1056 + c] = (_Float16)v;
  }
  for (int t = tid; t < 16 * 16; t += 128) {
    int e = t >> 4, c = 336 + (t & 15);
    sH[e * 352 + c] = (_Float16)0.f;
  }
  __syncthreads();

  const int wave = tid >> 5;
  const int lane = tid & 31;
  const int row  = lane & 15;
  const int aSel = (lane >> 4) << 3;
  const int bSel = (lane >> 4) << 3;
  const int rb   = (lane >> 4) << 3;

  for (int nt = wave; nt < 21; nt += 4) {  // 336/16 = 21 tiles
    const int col = (nt << 4) + row;
    const _Float16* aBase = &sA[row * 1056 + aSel];
    const _Float16* bBase = Wp1 + (col * 1056 + bSel);
    v8f acc = {0.f, 0.f, 0.f, 0.f, 0.f, 0.f, 0.f, 0.f};
    for (int kk = 0; kk < 1056; kk += 32) {
      v8h alo = *(const v8h*)(aBase + kk);
      v8h ahi = *(const v8h*)(aBase + kk + 16);
      v16h af = __builtin_shufflevector(alo, ahi, 0, 1, 2, 3, 4, 5, 6, 7, 8, 9,
                                        10, 11, 12, 13, 14, 15);
      v8h blo = *(const v8h*)(bBase + kk);
      v8h bhi = *(const v8h*)(bBase + kk + 16);
      v16h bf = __builtin_shufflevector(blo, bhi, 0, 1, 2, 3, 4, 5, 6, 7, 8, 9,
                                        10, 11, 12, 13, 14, 15);
      acc = __builtin_amdgcn_wmma_f32_16x16x32_f16(false, af, false, bf,
                                                   (short)0, acc, false, false);
    }
    const float bias = bn1[col];
#pragma unroll
    for (int r = 0; r < 8; ++r)
      sH[(rb + r) * 352 + col] = (_Float16)lrelu(acc[r] + bias);
  }
  __syncthreads();

  for (int nt = wave; nt < 16; nt += 4) {
    const int col = (nt << 4) + row;
    const _Float16* aBase = &sH[row * 352 + aSel];
    const _Float16* bBase = Wp2 + (col * 352 + bSel);
    v8f acc = {0.f, 0.f, 0.f, 0.f, 0.f, 0.f, 0.f, 0.f};
    for (int kk = 0; kk < 352; kk += 32) {
      v8h alo = *(const v8h*)(aBase + kk);
      v8h ahi = *(const v8h*)(aBase + kk + 16);
      v16h af = __builtin_shufflevector(alo, ahi, 0, 1, 2, 3, 4, 5, 6, 7, 8, 9,
                                        10, 11, 12, 13, 14, 15);
      v8h blo = *(const v8h*)(bBase + kk);
      v8h bhi = *(const v8h*)(bBase + kk + 16);
      v16h bf = __builtin_shufflevector(blo, bhi, 0, 1, 2, 3, 4, 5, 6, 7, 8, 9,
                                        10, 11, 12, 13, 14, 15);
      acc = __builtin_amdgcn_wmma_f32_16x16x32_f16(false, af, false, bf,
                                                   (short)0, acc, false, false);
    }
    const float bias = bn2[col];
#pragma unroll
    for (int r = 0; r < 8; ++r)
      x2[(size_t)(g0 + rb + r) * 256 + col] = acc[r] + bias;  // no activation
  }
}

// ---------------------------------------------------------------------------
// Pooling: per batch max/min/sum/mean over nodes + homophily + n_pulses,
// leaky applied to the whole pooled vector (matches reference).
// ---------------------------------------------------------------------------
__global__ __launch_bounds__(256) void pool_kernel(const float* __restrict__ x2,
                                                   const float* __restrict__ h,
                                                   const float* __restrict__ n_pulses,
                                                   float* __restrict__ pooled) {
  const int b = blockIdx.x;
  const int t = threadIdx.x;
  float mx = -3.0e38f, mn = 3.0e38f, sm = 0.f;
  for (int i = 0; i < NPER; ++i) {
    float v = x2[(size_t)((b << 8) + i) * 256 + t];
    mx = fmaxf(mx, v);
    mn = fminf(mn, v);
    sm += v;
  }
  float* p = pooled + (size_t)b * 1029;
  p[t]       = lrelu(mx);
  p[256 + t] = lrelu(mn);
  p[512 + t] = lrelu(sm);
  p[768 + t] = lrelu(sm * (1.f / 256.f));
  if (t == 0) {
    p[1024] = lrelu(h[b * 4 + 3]);
    p[1025] = lrelu(h[b * 4 + 0]);
    p[1026] = lrelu(h[b * 4 + 1]);
    p[1027] = lrelu(h[b * 4 + 2]);
    p[1028] = lrelu(n_pulses[b]);
  }
}

// ---------------------------------------------------------------------------
// Head: pooled(1029) @ Wn3(1029x128)+bn3 leaky -> @ Wn4(128x3)+bn4, tanh[0:2].
// ---------------------------------------------------------------------------
__global__ __launch_bounds__(128) void head_kernel(
    const float* __restrict__ pooled, const float* __restrict__ Wn3,
    const float* __restrict__ bn3, const float* __restrict__ Wn4,
    const float* __restrict__ bn4, float* __restrict__ out) {
  __shared__ float hs[128];
  const int b = blockIdx.x;
  const int t = threadIdx.x;
  const float* p = pooled + (size_t)b * 1029;
  float acc = bn3[t];
  for (int r = 0; r < 1029; ++r) acc += p[r] * Wn3[(size_t)r * 128 + t];
  hs[t] = lrelu(acc);
  __syncthreads();
  if (t < 3) {
    float o = bn4[t];
    for (int r = 0; r < 128; ++r) o += hs[r] * Wn4[r * 3 + t];
    if (t < 2) o = tanhf(o);
    out[b * 3 + t] = o;
  }
}

// ---------------------------------------------------------------------------
extern "C" void kernel_launch(void* const* d_in, const int* in_sizes, int n_in,
                              void* d_out, int out_size, void* d_ws, size_t ws_size,
                              hipStream_t stream) {
  (void)in_sizes; (void)n_in; (void)out_size; (void)ws_size;
  const float* x        = (const float*)d_in[0];
  const float* n_pulses = (const float*)d_in[1];
  const float* W1a = (const float*)d_in[2];  const float* b1a = (const float*)d_in[3];
  const float* W1b = (const float*)d_in[4];  const float* b1b = (const float*)d_in[5];
  const float* W2a = (const float*)d_in[6];  const float* b2a = (const float*)d_in[7];
  const float* W2b = (const float*)d_in[8];  const float* b2b = (const float*)d_in[9];
  const float* W3a = (const float*)d_in[10]; const float* b3a = (const float*)d_in[11];
  const float* W3b = (const float*)d_in[12]; const float* b3b = (const float*)d_in[13];
  const float* W4a = (const float*)d_in[14]; const float* b4a = (const float*)d_in[15];
  const float* W4b = (const float*)d_in[16]; const float* b4b = (const float*)d_in[17];
  const float* Wn1 = (const float*)d_in[18]; const float* bn1 = (const float*)d_in[19];
  const float* Wn2 = (const float*)d_in[20]; const float* bn2 = (const float*)d_in[21];
  const float* Wn3 = (const float*)d_in[22]; const float* bn3 = (const float*)d_in[23];
  const float* Wn4 = (const float*)d_in[24]; const float* bn4 = (const float*)d_in[25];

  char* w = (char*)d_ws;
  size_t off = 0;
  auto take = [&](size_t bytes) -> char* {
    char* p = w + off;
    off += (bytes + 255) & ~(size_t)255;
    return p;
  };
  const size_t FEATB = (size_t)NBATCH * NPER * 256 * sizeof(float);  // 16 MB
  int*   idxp   = (int*)take((size_t)NBATCH * NPER * KNN * sizeof(int));
  float* fa     = (float*)take(FEATB);
  float* fbuf   = (float*)take(FEATB);
  float* fcbuf  = (float*)take(FEATB);
  float* fdbuf  = (float*)take(FEATB);
  float* x2buf  = (float*)take(FEATB);
  float* hbuf   = (float*)take((size_t)NBATCH * 4 * sizeof(float));
  float* pooled = (float*)take((size_t)NBATCH * 1029 * sizeof(float));
  _Float16* W1ap = (_Float16*)take((size_t)128 * 32   * 2);
  _Float16* W1bp = (_Float16*)take((size_t)256 * 128  * 2);
  _Float16* W2ap = (_Float16*)take((size_t)336 * 512  * 2);
  _Float16* W2bp = (_Float16*)take((size_t)256 * 352  * 2);
  _Float16* W3ap = (_Float16*)take((size_t)336 * 512  * 2);
  _Float16* W3bp = (_Float16*)take((size_t)256 * 352  * 2);
  _Float16* W4ap = (_Float16*)take((size_t)336 * 512  * 2);
  _Float16* W4bp = (_Float16*)take((size_t)256 * 352  * 2);
  _Float16* Wn1p = (_Float16*)take((size_t)336 * 1056 * 2);
  _Float16* Wn2p = (_Float16*)take((size_t)256 * 352  * 2);

  auto packGrid = [](int n, int kpad) { return (n * kpad + 255) / 256; };
  pack_weight_kernel<<<packGrid(128, 32), 256, 0, stream>>>(W1a, 14, 128, 32, W1ap);
  pack_weight_kernel<<<packGrid(256, 128), 256, 0, stream>>>(W1b, 128, 256, 128, W1bp);
  pack_weight_kernel<<<packGrid(336, 512), 256, 0, stream>>>(W2a, 512, 336, 512, W2ap);
  pack_weight_kernel<<<packGrid(256, 352), 256, 0, stream>>>(W2b, 336, 256, 352, W2bp);
  pack_weight_kernel<<<packGrid(336, 512), 256, 0, stream>>>(W3a, 512, 336, 512, W3ap);
  pack_weight_kernel<<<packGrid(256, 352), 256, 0, stream>>>(W3b, 336, 256, 352, W3bp);
  pack_weight_kernel<<<packGrid(336, 512), 256, 0, stream>>>(W4a, 512, 336, 512, W4ap);
  pack_weight_kernel<<<packGrid(256, 352), 256, 0, stream>>>(W4b, 336, 256, 352, W4bp);
  pack_weight_kernel<<<packGrid(336, 1056), 256, 0, stream>>>(Wn1, 1031, 336, 1056, Wn1p);
  pack_weight_kernel<<<packGrid(256, 352), 256, 0, stream>>>(Wn2, 336, 256, 352, Wn2p);

  const int nEdgeBlocks = NBATCH * NPER / 2;  // 8192
  const int nNodeBlocks = NBATCH * NPER / 16; // 1024

  // Layer 1: KNN on x coords, homophily, EdgeConv(7 -> 128 -> 256)
  knn_kernel<<<NBATCH, NPER, 0, stream>>>(x, 7, idxp);
  homophily_kernel<<<NBATCH, NPER, 0, stream>>>(x, idxp, hbuf);
  edgeconv_kernel<<<nEdgeBlocks, 128, 0, stream>>>(x, 7, 7, 32, idxp,
                                                   W1ap, b1a, 128, 128,
                                                   W1bp, b1b, fa);
  // Layer 2
  knn_kernel<<<NBATCH, NPER, 0, stream>>>(fa, 256, idxp);
  edgeconv_kernel<<<nEdgeBlocks, 128, 0, stream>>>(fa, 256, 256, 512, idxp,
                                                   W2ap, b2a, 336, 352,
                                                   W2bp, b2b, fbuf);
  // Layer 3
  knn_kernel<<<NBATCH, NPER, 0, stream>>>(fbuf, 256, idxp);
  edgeconv_kernel<<<nEdgeBlocks, 128, 0, stream>>>(fbuf, 256, 256, 512, idxp,
                                                   W3ap, b3a, 336, 352,
                                                   W3bp, b3b, fcbuf);
  // Layer 4
  knn_kernel<<<NBATCH, NPER, 0, stream>>>(fcbuf, 256, idxp);
  edgeconv_kernel<<<nEdgeBlocks, 128, 0, stream>>>(fcbuf, 256, 256, 512, idxp,
                                                   W4ap, b4a, 336, 352,
                                                   W4bp, b4b, fdbuf);
  // Node MLP on concat features
  nodemlp_kernel<<<nNodeBlocks, 128, 0, stream>>>(x, fa, fbuf, fcbuf, fdbuf,
                                                  Wn1p, bn1, Wn2p, bn2, x2buf);
  // Pooling + head
  pool_kernel<<<NBATCH, NPER, 0, stream>>>(x2buf, hbuf, n_pulses, pooled);
  head_kernel<<<NBATCH, 128, 0, stream>>>(pooled, Wn3, bn3, Wn4, bn4,
                                          (float*)d_out);
}
